// DenseAttentionLayer_3788161155599
// MI455X (gfx1250) — matrix operand order repaired and checked by reading, not compiled, not run on hardware
//
#include <hip/hip_runtime.h>
#include <hip/hip_bf16.h>
#include <stdint.h>

#define N_REL  8192
#define IN_DIM 512
#define HID    256
#define REL_DIM 256

typedef __attribute__((ext_vector_type(16))) __bf16 v16bf;
typedef __attribute__((ext_vector_type(8)))  float  v8f;

struct __attribute__((aligned(16))) U4 { unsigned int x, y, z, w; };
struct __attribute__((aligned(8)))  U2 { unsigned int x, y; };
union Frag { v16bf v; U4 q[2]; };

__device__ __forceinline__ unsigned int f2bf1(float f) {
  unsigned int u = __float_as_uint(f);
  return (u + 0x7FFFu + ((u >> 16) & 1u)) >> 16;   // RNE f32 -> bf16
}
__device__ __forceinline__ unsigned int pack_bf2(float lo, float hi) {
#if __has_builtin(__builtin_amdgcn_cvt_pk_bf16_f32)
  auto p = __builtin_amdgcn_cvt_pk_bf16_f32(lo, hi);
  union C { decltype(p) v; unsigned int u; } c;
  c.v = p;
  return c.u;
#else
  return f2bf1(lo) | (f2bf1(hi) << 16);
#endif
}
__device__ __forceinline__ float leaky02(float v) { return v >= 0.f ? v : 0.2f * v; }

// ---------------------------------------------------------------------------
// 1) u_left[k] = sum_h W[h,k]*w_left[h]  (and u_right): 512 threads, 1 block
// ---------------------------------------------------------------------------
__global__ __launch_bounds__(IN_DIM)
void k_prep_u(const float* __restrict__ W, const float* __restrict__ wl,
              const float* __restrict__ wr, float* __restrict__ u) {
  int k = threadIdx.x;
  float sl = 0.f, sr = 0.f;
  for (int h = 0; h < HID; ++h) {
    float w = W[h * IN_DIM + k];
    sl = fmaf(w, wl[h], sl);
    sr = fmaf(w, wr[h], sr);
  }
  u[k] = sl;
  u[IN_DIM + k] = sr;
}

// ---------------------------------------------------------------------------
// 2) left[i] = context[i,:].u_left + b_left ; right likewise. One wave32/row.
// ---------------------------------------------------------------------------
__global__ __launch_bounds__(256)
void k_prep_lr(const float* __restrict__ ctx, const float* __restrict__ u,
               const float* __restrict__ bl, const float* __restrict__ br,
               float* __restrict__ leftv, float* __restrict__ rightv) {
  int row  = (int)((blockIdx.x * blockDim.x + threadIdx.x) >> 5);
  int lane = threadIdx.x & 31;
  if (row >= N_REL) return;
  const float* c = ctx + (size_t)row * IN_DIM;
  float al = 0.f, ar = 0.f;
  for (int k = lane; k < IN_DIM; k += 32) {
    float cv = c[k];
    al = fmaf(cv, u[k], al);
    ar = fmaf(cv, u[IN_DIM + k], ar);
  }
  for (int off = 16; off; off >>= 1) {
    al += __shfl_xor(al, off);
    ar += __shfl_xor(ar, off);
  }
  if (lane == 0) {
    leftv[row]  = al + bl[0];
    rightv[row] = ar + br[0];
  }
}

// ---------------------------------------------------------------------------
// 3) relT[n, j] = bf16(relation[j, n])  (256 x 8192 bf16, row-major)
// ---------------------------------------------------------------------------
__global__ __launch_bounds__(256)
void k_transpose_rel(const float* __restrict__ rel, unsigned short* __restrict__ relT) {
  __shared__ float tile[32][33];
  int j0 = blockIdx.x * 32;
  int n0 = blockIdx.y * 32;
  int tx = threadIdx.x & 31;
  int ty = threadIdx.x >> 5;               // 0..7
  for (int yy = ty; yy < 32; yy += 8)
    tile[yy][tx] = rel[(size_t)(j0 + yy) * REL_DIM + n0 + tx];
  __syncthreads();
  for (int yy = ty; yy < 32; yy += 8)
    relT[(size_t)(n0 + yy) * N_REL + j0 + tx] = (unsigned short)f2bf1(tile[tx][yy]);
}

// ---------------------------------------------------------------------------
// 4) Per-row masked online softmax stats: M[i] = max, Zinv[i] = 1/sum.
//    One wave32 per row, streaming adj (pure bandwidth).
// ---------------------------------------------------------------------------
__global__ __launch_bounds__(256)
void k_row_stats(const float* __restrict__ adj, const float* __restrict__ leftv,
                 const float* __restrict__ rightv, float* __restrict__ Mv,
                 float* __restrict__ Zinv) {
  int row  = (int)((blockIdx.x * blockDim.x + threadIdx.x) >> 5);
  int lane = threadIdx.x & 31;
  if (row >= N_REL) return;
  const float* arow = adj + (size_t)row * N_REL;
  const float li = leftv[row];
  float m = -1e30f, s = 0.f;
  for (int j = lane; j < N_REL; j += 32) {
    if (j + 4096 < N_REL) __builtin_prefetch(arow + j + 4096, 0, 1);
    float a = arow[j];
    float v = leaky02(li + rightv[j]);
    if (a > 0.f) {
      if (v > m) { s = s * __expf(m - v) + 1.f; m = v; }
      else       { s += __expf(v - m); }
    }
  }
  // wave-wide online-softmax merge
  for (int off = 16; off; off >>= 1) {
    float m2 = __shfl_xor(m, off);
    float s2 = __shfl_xor(s, off);
    float mn = fmaxf(m, m2);
    s = s * __expf(m - mn) + s2 * __expf(m2 - mn);
    m = mn;
  }
  if (lane == 0) { Mv[row] = m; Zinv[row] = 1.f / s; }
}

// ---------------------------------------------------------------------------
// 5) x = relu( (P @ relation) * Zinv )  via v_wmma_f32_16x16x32_bf16.
//    Workgroup: 512 thr = 16 waves = 2 row-tiles x 8 col-groups; each wave
//    owns one 16-row group and two 16-col tiles (32 cols). adj read once.
//    K=64 per iteration, double-buffered LDS weight tile -> ONE barrier/iter.
// ---------------------------------------------------------------------------
__global__ __launch_bounds__(512)
void k_attn_gemm(const float* __restrict__ adj, const float* __restrict__ leftv,
                 const float* __restrict__ rightv, const float* __restrict__ Mv,
                 const float* __restrict__ Zinv,
                 const unsigned short* __restrict__ relT, float* __restrict__ out) {
  constexpr int ROWS    = 32;            // rows per workgroup
  constexpr int KC      = 64;            // K per iteration (2 WMMA chunks)
  constexpr int LSTRIDE = 36;            // dwords/row: 32 data (64 bf16) + 4 pad
  constexpr int BUFSZ   = ROWS * LSTRIDE;
  __shared__ __align__(16) unsigned int w_lds[2 * BUFSZ];

  const int i0   = blockIdx.x * ROWS;
  const int tid  = threadIdx.x;
  const int lane = tid & 31;
  const int wave = tid >> 5;
  const int row_group = wave >> 3;  // 0..1
  const int col_group = wave & 7;   // 0..7

  // --- producer role: each thread makes 4 weights (2 packed dwords) per iter
  const int wr = tid >> 4;          // 0..31: row within stripe
  const int wt = tid & 15;          // k quad index
  const float rowL = leftv[i0 + wr];
  const float rowM = Mv[i0 + wr];
  const float* adj_row = adj + (size_t)(i0 + wr) * (size_t)N_REL;
  const unsigned int wslot = (unsigned)wr * LSTRIDE + 2 * wt;

  // --- consumer A-frag offsets (16-bit A 16x32 layout, ISA 7.12.2)
  const int m  = lane & 15;
  const int hi = lane >> 4;
  const unsigned int a_base = (unsigned)(row_group * 16 + m) * LSTRIDE + hi * 4;

  // B-fragment: lane holds column n (relT row), lanes<16: K0..7,16..23, lanes>=16: K8..15,24..31
  const int ntile0 = col_group * 32;
  const unsigned short* b0p = relT + (size_t)(ntile0 + m)      * N_REL + hi * 8;
  const unsigned short* b1p = relT + (size_t)(ntile0 + 16 + m) * N_REL + hi * 8;

  v8f acc0 = {0.f, 0.f, 0.f, 0.f, 0.f, 0.f, 0.f, 0.f};
  v8f acc1 = acc0;

  for (int k0 = 0; k0 < N_REL; k0 += KC) {
    const unsigned int pbase = ((k0 >> 6) & 1) ? BUFSZ : 0u;

    // produce P tile: w = mask * exp(leaky(left+right) - M), fp32 -> bf16
    const float4 av = *(const float4*)(adj_row + k0 + 4 * wt);
    const float4 rv = *(const float4*)(rightv + k0 + 4 * wt);
    if (k0 + 2048 < N_REL) __builtin_prefetch(adj_row + k0 + 2048, 0, 1);
    float w0 = (av.x > 0.f) ? __expf(leaky02(rowL + rv.x) - rowM) : 0.f;
    float w1 = (av.y > 0.f) ? __expf(leaky02(rowL + rv.y) - rowM) : 0.f;
    float w2 = (av.z > 0.f) ? __expf(leaky02(rowL + rv.z) - rowM) : 0.f;
    float w3 = (av.w > 0.f) ? __expf(leaky02(rowL + rv.w) - rowM) : 0.f;
    U2 packed;
    packed.x = pack_bf2(w0, w1);
    packed.y = pack_bf2(w2, w3);
    *(U2*)&w_lds[pbase + wslot] = packed;

    __syncthreads();   // single barrier: double-buffer makes writes of the
                       // *other* buffer safe until the next barrier

    Frag a, b;
    // ---- K chunk 0 (k0 .. k0+31)
    a.q[0] = *(const U4*)&w_lds[pbase + a_base];
    a.q[1] = *(const U4*)&w_lds[pbase + a_base + 8];
    b.q[0] = *(const U4*)(b0p + k0);
    b.q[1] = *(const U4*)(b0p + k0 + 16);
    acc0 = __builtin_amdgcn_wmma_f32_16x16x32_bf16(false, a.v, false, b.v,
                                                   (short)0, acc0, false, false);
    b.q[0] = *(const U4*)(b1p + k0);
    b.q[1] = *(const U4*)(b1p + k0 + 16);
    acc1 = __builtin_amdgcn_wmma_f32_16x16x32_bf16(false, a.v, false, b.v,
                                                   (short)0, acc1, false, false);
    // ---- K chunk 1 (k0+32 .. k0+63)
    a.q[0] = *(const U4*)&w_lds[pbase + a_base + 16];
    a.q[1] = *(const U4*)&w_lds[pbase + a_base + 24];
    b.q[0] = *(const U4*)(b0p + k0 + 32);
    b.q[1] = *(const U4*)(b0p + k0 + 48);
    acc0 = __builtin_amdgcn_wmma_f32_16x16x32_bf16(false, a.v, false, b.v,
                                                   (short)0, acc0, false, false);
    b.q[0] = *(const U4*)(b1p + k0 + 32);
    b.q[1] = *(const U4*)(b1p + k0 + 48);
    acc1 = __builtin_amdgcn_wmma_f32_16x16x32_bf16(false, a.v, false, b.v,
                                                   (short)0, acc1, false, false);
  }

  // epilogue: scale by 1/Z, ReLU, store. C layout: VGPR v -> M = hi*8+v, N = lane&15
  const int base_row = i0 + row_group * 16 + hi * 8;
#pragma unroll
  for (int v = 0; v < 8; ++v) {
    int row = base_row + v;
    float z  = Zinv[row];
    float x0 = acc0[v] * z;
    float x1 = acc1[v] * z;
    out[(size_t)row * REL_DIM + ntile0 + m]      = x0 > 0.f ? x0 : 0.f;
    out[(size_t)row * REL_DIM + ntile0 + 16 + m] = x1 > 0.f ? x1 : 0.f;
  }
}

// ---------------------------------------------------------------------------
extern "C" void kernel_launch(void* const* d_in, const int* in_sizes, int n_in,
                              void* d_out, int out_size, void* d_ws, size_t ws_size,
                              hipStream_t stream) {
  (void)in_sizes; (void)n_in; (void)out_size; (void)ws_size;
  const float* relation = (const float*)d_in[0];
  const float* context  = (const float*)d_in[1];
  const float* adj      = (const float*)d_in[2];
  const float* Wc       = (const float*)d_in[3];
  const float* wl       = (const float*)d_in[4];
  const float* bl       = (const float*)d_in[5];
  const float* wr       = (const float*)d_in[6];
  const float* br       = (const float*)d_in[7];
  float* out = (float*)d_out;

  char* ws = (char*)d_ws;
  float* u      = (float*)(ws);                          // 1024 floats
  float* leftv  = (float*)(ws + 4096);                   // 8192 floats
  float* rightv = (float*)(ws + 4096 + 1 * 32768);       // 8192 floats
  float* Mv     = (float*)(ws + 4096 + 2 * 32768);       // 8192 floats
  float* Zinv   = (float*)(ws + 4096 + 3 * 32768);       // 8192 floats
  unsigned short* relT = (unsigned short*)(ws + 4096 + 4 * 32768);  // 256*8192 bf16

  k_prep_u<<<1, IN_DIM, 0, stream>>>(Wc, wl, wr, u);
  k_prep_lr<<<(N_REL * 32) / 256, 256, 0, stream>>>(context, u, bl, br, leftv, rightv);
  k_transpose_rel<<<dim3(N_REL / 32, REL_DIM / 32), 256, 0, stream>>>(relation, relT);
  k_row_stats<<<(N_REL * 32) / 256, 256, 0, stream>>>(adj, leftv, rightv, Mv, Zinv);
  k_attn_gemm<<<N_REL / 32, 512, 0, stream>>>(adj, leftv, rightv, Mv, Zinv, relT, out);
}
